// TemporalAttn_59253368815798
// MI455X (gfx1250) — matrix operand order repaired and checked by reading, not compiled, not run on hardware
//
#include <hip/hip_runtime.h>
#include <hip/hip_bf16.h>

typedef unsigned short u16;
typedef __attribute__((ext_vector_type(16))) __bf16 v16bf;
typedef __attribute__((ext_vector_type(8)))  __bf16 v8bf;
typedef __attribute__((ext_vector_type(8)))  float  v8f;
typedef int v4i_g __attribute__((vector_size(16)));   // matches builtin proto

union AFrag { v16bf v; v8bf h[2]; };

#define TB 4
#define TT 1024
#define TC 16
#define TD 256
#define NROW (TB*TT*TC)   // 65536 flattened (b,t,c) rows

#if defined(__gfx1250__) && __has_builtin(__builtin_amdgcn_global_load_async_to_lds_b128) && __has_builtin(__builtin_amdgcn_s_wait_asynccnt)
#define HAVE_ASYNC 1
#define ASYNC_WAIT(n) __builtin_amdgcn_s_wait_asynccnt(n)
#else
#define HAVE_ASYNC 0
#define ASYNC_WAIT(n)
#endif

__device__ __forceinline__ u16 f2bf(float f) {
  unsigned u = __builtin_bit_cast(unsigned, f);
  u += 0x7fffu + ((u >> 16) & 1u);
  return (u16)(u >> 16);
}

// 16-byte global -> LDS copy (async on CDNA5 if available)
__device__ __forceinline__ void cp16(const u16* g, u16* l) {
#if HAVE_ASYNC
  __builtin_amdgcn_global_load_async_to_lds_b128(
      (__attribute__((address_space(1))) v4i_g*)g,
      (__attribute__((address_space(3))) v4i_g*)l, 0, 0);
#else
  *(v8bf*)l = *(const v8bf*)g;
#endif
}

// ---------------- K0: cast x and the three weight matrices to bf16 ----------
__global__ void __launch_bounds__(256)
k_cast(const float* __restrict__ x,  const float* __restrict__ wq,
       const float* __restrict__ wk, const float* __restrict__ wv,
       u16* __restrict__ xh, u16* __restrict__ wqh,
       u16* __restrict__ wkh, u16* __restrict__ wvh) {
  long i = (long)blockIdx.x * blockDim.x + threadIdx.x;
  const long nx = (long)NROW * TD;
  const long nw = (long)TD * TD;
  if (i < nx)        { xh[i]  = f2bf(x[i]);  return; }
  long j = i - nx;
  if (j < nw)        { wqh[j] = f2bf(wq[j]); return; }
  j -= nw;
  if (j < nw)        { wkh[j] = f2bf(wk[j]); return; }
  j -= nw;
  if (j < nw)        { wvh[j] = f2bf(wv[j]); }
}

// ---------------- K1: QKV projection (WMMA) + bias + RoPE + layout ----------
__global__ void __launch_bounds__(128)
k_qkv(const u16* __restrict__ xh,
      const u16* __restrict__ wqh, const u16* __restrict__ wkh, const u16* __restrict__ wvh,
      const float* __restrict__ bq, const float* __restrict__ bk, const float* __restrict__ bv,
      const int* __restrict__ pos, const float* __restrict__ pe,
      u16* __restrict__ qh, u16* __restrict__ kh, u16* __restrict__ vT) {
  const int lane = threadIdx.x & 31;
  const int wid  = blockIdx.x * 4 + (threadIdx.x >> 5);
  const int mt   = wid >> 4;          // 0..4095 row tiles
  const int nt   = wid & 15;          // 0..15  outdim tiles
  const int half = lane >> 4, lm = lane & 15;
  const int arow = mt*16 + lm;
  const int n    = nt*16 + lm;

  v8f cq = {}, ck = {}, cv = {};
  const u16* xrow = xh  + (long)arow * TD;
  const u16* wqr  = wqh + (long)n * TD;
  const u16* wkr  = wkh + (long)n * TD;
  const u16* wvr  = wvh + (long)n * TD;
  #pragma unroll
  for (int ks = 0; ks < TD; ks += 32) {
    AFrag a;
    a.h[0] = *(const v8bf*)(xrow + ks + 16*half);
    a.h[1] = *(const v8bf*)(xrow + ks + 16*half + 8);
    v16bf bqf = *(const v16bf*)(wqr + ks + 16*half);
    v16bf bkf = *(const v16bf*)(wkr + ks + 16*half);
    v16bf bvf = *(const v16bf*)(wvr + ks + 16*half);
    cq = __builtin_amdgcn_wmma_f32_16x16x32_bf16(false, a.v, false, bqf, (short)0, cq, false, false);
    ck = __builtin_amdgcn_wmma_f32_16x16x32_bf16(false, a.v, false, bkf, (short)0, ck, false, false);
    cv = __builtin_amdgcn_wmma_f32_16x16x32_bf16(false, a.v, false, bvf, (short)0, cv, false, false);
  }
  const float biasq = bq[n], biask = bk[n], biasv = bv[n];
  const int  j   = n >> 1;
  const bool odd = (n & 1) != 0;
  #pragma unroll
  for (int i = 0; i < 8; ++i) {
    int r  = mt*16 + i + 8*half;
    int b  = r >> 14;
    int t  = (r >> 4) & (TT-1);
    int c  = r & (TC-1);
    int bc = b*TC + c;
    int p  = pos[r];
    float co = pe[((long)p*(TD/2) + j)*2 + 0];
    float si = pe[((long)p*(TD/2) + j)*2 + 1];
    float qv = cq[i] + biasq;
    float qp = __shfl_xor(qv, 1);
    float qr = odd ? (qp*si + qv*co) : (qv*co - qp*si);
    float kv = ck[i] + biask;
    float kp = __shfl_xor(kv, 1);
    float kr = odd ? (kp*si + kv*co) : (kv*co - kp*si);
    long qaddr = ((long)(bc*TT + t))*TD + n;
    qh[qaddr] = f2bf(qr);
    kh[qaddr] = f2bf(kr);
    vT[((long)(bc*TD + n))*TT + t] = f2bf(cv[i] + biasv);
  }
}

// ---------------- K3: flash attention, double-buffered async LDS staging ----
__global__ void __launch_bounds__(128)
k_attn(const u16* __restrict__ qh, const u16* __restrict__ kh, const u16* __restrict__ vT,
       const unsigned char* __restrict__ xmask, float* __restrict__ of) {
  __shared__ __align__(32) u16 kbuf[2][32][TD];   // 2 x 16 KB : K rows of chunk
  __shared__ __align__(32) u16 vbuf[2][TD][32];   // 2 x 16 KB : V^T rows of chunk
  __shared__ __align__(32) u16 pbuf[4][16*32];    // P tile [q][key] per wave
  __shared__ float rbuf[4][16], mbuf[4][16], sbuf[4][16];

  const int tid  = threadIdx.x;
  const int lane = tid & 31;
  const int w    = tid >> 5;
  const int wid  = blockIdx.x*4 + w;
  const int bc   = wid >> 6;            // same for all 4 waves of a block
  const int qt   = wid & 63;
  const int b    = bc >> 4, c = bc & 15;
  const int half = lane >> 4, lm = lane & 15;

  const u16* kbase = kh + (long)bc*TT*TD;
  const u16* vbase = vT + (long)bc*TD*TT;

  // cooperative stage of one 32-key chunk (K: 32x256, V^T: 256x32) into LDS
  auto stage = [&](int ch, int buf) {
    #pragma unroll
    for (int it = 0; it < 8; ++it) {          // K: 1024 x 16B segments
      int e   = tid + 128*it;
      int row = e >> 5, seg = e & 31;
      cp16(kbase + (long)(ch + row)*TD + seg*8, &kbuf[buf][row][seg*8]);
    }
    #pragma unroll
    for (int it = 0; it < 8; ++it) {          // V: 1024 x 16B segments
      int e   = tid + 128*it;
      int row = e >> 2, seg = e & 3;
      cp16(vbase + (long)row*TT + ch + seg*8, &vbuf[buf][row][seg*8]);
    }
  };

  // Q A-fragments held in VGPRs for the whole key loop
  AFrag qf[8];
  const u16* qrow = qh + ((long)(bc*TT + qt*16 + lm))*TD;
  #pragma unroll
  for (int dch = 0; dch < 8; ++dch) {
    qf[dch].h[0] = *(const v8bf*)(qrow + dch*32 + 16*half);
    qf[dch].h[1] = *(const v8bf*)(qrow + dch*32 + 16*half + 8);
  }

  v8f oacc[16];
  const v8f vz = {};
  #pragma unroll
  for (int i = 0; i < 16; ++i) oacc[i] = vz;
  float m[8], ssum[8];
  #pragma unroll
  for (int i = 0; i < 8; ++i) { m[i] = -3e38f; ssum[i] = 0.f; }

  const float scale = 0.0625f;          // 1/sqrt(256)
  stage(0, 0);
  int cur = 0;
  for (int ch = 0; ch < TT; ch += 32) {
    if (ch + 32 < TT) { stage(ch + 32, cur ^ 1); ASYNC_WAIT(16); }
    else              { ASYNC_WAIT(0); }
    __syncthreads();                    // chunk `ch` resident in kbuf/vbuf[cur]

    v8f s0 = vz, s1 = vz;
    #pragma unroll
    for (int dch = 0; dch < 8; ++dch) {
      v16bf b0 = *(const v16bf*)(&kbuf[cur][lm     ][dch*32 + 16*half]);
      v16bf b1 = *(const v16bf*)(&kbuf[cur][16 + lm][dch*32 + 16*half]);
      s0 = __builtin_amdgcn_wmma_f32_16x16x32_bf16(false, qf[dch].v, false, b0, (short)0, s0, false, false);
      s1 = __builtin_amdgcn_wmma_f32_16x16x32_bf16(false, qf[dch].v, false, b1, (short)0, s1, false, false);
    }
    bool km0 = xmask[(b*TT + ch +      lm)*TC + c] != 0;
    bool km1 = xmask[(b*TT + ch + 16 + lm)*TC + c] != 0;
    #pragma unroll
    for (int i = 0; i < 8; ++i) {
      float a0 = km0 ? -1e30f : s0[i]*scale;
      float a1 = km1 ? -1e30f : s1[i]*scale;
      float mx = fmaxf(a0, a1);
      #pragma unroll
      for (int off = 1; off < 16; off <<= 1) mx = fmaxf(mx, __shfl_xor(mx, off));
      float mn = fmaxf(m[i], mx);
      float r  = __expf(m[i] - mn);
      float p0 = __expf(a0 - mn);
      float p1 = __expf(a1 - mn);
      float rs = p0 + p1;
      #pragma unroll
      for (int off = 1; off < 16; off <<= 1) rs += __shfl_xor(rs, off);
      ssum[i] = ssum[i]*r + rs;
      m[i]    = mn;
      int q = i + 8*half;
      pbuf[w][q*32 +      lm] = f2bf(p0);
      pbuf[w][q*32 + 16 + lm] = f2bf(p1);
      if (lm == 0) rbuf[w][q] = r;
    }
    __syncthreads();
    float rq = rbuf[w][lm];
    v16bf pT = *(const v16bf*)(&pbuf[w][lm*32 + 16*half]);   // P^T B-fragment
    #pragma unroll
    for (int dt = 0; dt < 16; ++dt) {
      #pragma unroll
      for (int i = 0; i < 8; ++i) oacc[dt][i] *= rq;
      AFrag va;                                              // V^T A-fragment
      va.h[0] = *(const v8bf*)(&vbuf[cur][dt*16 + lm][16*half]);
      va.h[1] = *(const v8bf*)(&vbuf[cur][dt*16 + lm][16*half + 8]);
      oacc[dt] = __builtin_amdgcn_wmma_f32_16x16x32_bf16(false, va.v, false, pT, (short)0, oacc[dt], false, false);
    }
    __syncthreads();                    // protect pbuf + kbuf/vbuf[cur^1] reuse
    cur ^= 1;
  }
  if (lm == 0) {
    #pragma unroll
    for (int i = 0; i < 8; ++i) { mbuf[w][i+8*half] = m[i]; sbuf[w][i+8*half] = ssum[i]; }
  }
  __syncthreads();
  float mq = mbuf[w][lm], sq = sbuf[w][lm];
  bool qm = xmask[(b*TT + qt*16 + lm)*TC + c] != 0;
  float inv = (!qm && mq > -1e29f && sq > 0.f) ? 1.f/sq : 0.f;
  float* orow = of + ((long)(bc*TT + qt*16 + lm))*TD;
  #pragma unroll
  for (int dt = 0; dt < 16; ++dt)
    #pragma unroll
    for (int i = 0; i < 8; ++i)
      orow[dt*16 + i + 8*half] = oacc[dt][i]*inv;
}

// ---------------- K4: residual + layer-scale + LayerNorm --------------------
__global__ void __launch_bounds__(256)
k_ln(const float* __restrict__ x, const float* __restrict__ of,
     const float* __restrict__ lsg, const float* __restrict__ g,
     const float* __restrict__ bb, float* __restrict__ out) {
  __shared__ float red[256];
  __shared__ float red2[256];
  int r = blockIdx.x;
  int d = threadIdx.x;
  int b = r >> 14, t = (r >> 4) & (TT-1), c = r & 15;
  int bc = b*TC + c;
  float y = x[(long)r*TD + d] + lsg[d]*of[((long)(bc*TT + t))*TD + d];
  red[d] = y; red2[d] = y*y;
  __syncthreads();
  for (int s = 128; s > 0; s >>= 1) {
    if (d < s) { red[d] += red[d+s]; red2[d] += red2[d+s]; }
    __syncthreads();
  }
  float mu  = red[0] * (1.0f/TD);
  float var = red2[0]*(1.0f/TD) - mu*mu;
  out[(long)r*TD + d] = (y - mu) * rsqrtf(var + 1e-5f) * g[d] + bb[d];
}

extern "C" void kernel_launch(void* const* d_in, const int* in_sizes, int n_in,
                              void* d_out, int out_size, void* d_ws, size_t ws_size,
                              hipStream_t stream) {
  const float*         x     = (const float*)d_in[0];
  const unsigned char* xmask = (const unsigned char*)d_in[1];
  const int*           pos   = (const int*)d_in[2];
  const float*         pe    = (const float*)d_in[3];
  const float*         wq    = (const float*)d_in[4];
  const float*         bq    = (const float*)d_in[5];
  const float*         wk    = (const float*)d_in[6];
  const float*         bk    = (const float*)d_in[7];
  const float*         wv    = (const float*)d_in[8];
  const float*         bv    = (const float*)d_in[9];
  const float*         lng   = (const float*)d_in[10];
  const float*         lnb   = (const float*)d_in[11];
  const float*         lsg   = (const float*)d_in[12];

  char* ws = (char*)d_ws;
  size_t off = 0;
  u16* xh  = (u16*)(ws + off); off += (size_t)NROW*TD*2;
  u16* wqh = (u16*)(ws + off); off += (size_t)TD*TD*2;
  u16* wkh = (u16*)(ws + off); off += (size_t)TD*TD*2;
  u16* wvh = (u16*)(ws + off); off += (size_t)TD*TD*2;
  u16* qh  = (u16*)(ws + off); off += (size_t)NROW*TD*2;
  u16* kh  = (u16*)(ws + off); off += (size_t)NROW*TD*2;
  u16* vT  = (u16*)(ws + off); off += (size_t)NROW*TD*2;
  float* of = (float*)(ws + off); off += (size_t)NROW*TD*4;

  long ncast = (long)NROW*TD + 3L*TD*TD;
  k_cast<<<(int)((ncast + 255)/256), 256, 0, stream>>>(x, wq, wk, wv, xh, wqh, wkh, wvh);
  k_qkv <<<16384, 128, 0, stream>>>(xh, wqh, wkh, wvh, bq, bk, bv, pos, pe, qh, kh, vT);
  k_attn<<<1024, 128, 0, stream>>>(qh, kh, vT, xmask, of);
  k_ln  <<<NROW, 256, 0, stream>>>(x, of, lsg, lng, lnb, (float*)d_out);
}
